// ImprovedGNNClassifier_49314814493137
// MI455X (gfx1250) — compile-verified, matrix operand order checked by reading, not run on hardware
//
#include <hip/hip_runtime.h>
#include <hip/hip_bf16.h>

#define N_NODES 100000
#define E_EDGES 1600000
#define G_GRAPHS 64
#define H_DIM 128
#define BN_EPS 1e-5f

typedef __attribute__((ext_vector_type(16))) _Float16 v16h;
typedef __attribute__((ext_vector_type(8)))  _Float16 v8h;
typedef __attribute__((ext_vector_type(8)))  float    v8f;
typedef __attribute__((ext_vector_type(4)))  unsigned int v4u;
typedef __attribute__((ext_vector_type(8)))  int v8i;
typedef __attribute__((ext_vector_type(4)))  int v4i;

union AFrag { v16h v; v8h h[2]; };

// LDS row stride for Bt: 128 halves + 8 halves TDM pad = 272B = 68 DW
// -> lane l fragment reads start at bank 4*l mod 64: conflict-free b128 phases.
#define LDS_ROW_H 136
#define LDS_BYTES (H_DIM * LDS_ROW_H * 2)   // 34816 B

// ---------------- degree / norm ----------------
__global__ void k_init_deg(float* deg) {
    int i = blockIdx.x * blockDim.x + threadIdx.x;
    if (i < N_NODES) deg[i] = 1.0f;  // self-loop contributes 1 to every node's degree
}

__global__ void k_deg_edges(const int* __restrict__ ei, float* deg) {
    int e = blockIdx.x * blockDim.x + threadIdx.x;
    if (e < E_EDGES) atomicAdd(deg + ei[E_EDGES + e], 1.0f);  // dst row
}

__global__ void k_dinv(const float* __restrict__ deg, float* dinv) {
    int i = blockIdx.x * blockDim.x + threadIdx.x;
    if (i < N_NODES) {
        float d = deg[i];
        dinv[i] = (d > 0.0f) ? rsqrtf(d) : 0.0f;
    }
}

// ---------------- layer-1 GEMM (K=3, scalar) ----------------
__global__ void k_gemm_in(const float* __restrict__ x, const float* __restrict__ W1,
                          float* __restrict__ t) {
    int idx = blockIdx.x * blockDim.x + threadIdx.x;
    if (idx >= N_NODES * H_DIM) return;
    int i = idx >> 7, f = idx & (H_DIM - 1);
    const float* xi = x + (size_t)i * 3;
    t[idx] = xi[0] * W1[f] + xi[1] * W1[H_DIM + f] + xi[2] * W1[2 * H_DIM + f];
}

// ---------------- weight transpose + f16 convert ----------------
__global__ void k_wt16(const float* __restrict__ W, _Float16* __restrict__ Wt) {
    int idx = blockIdx.x * blockDim.x + threadIdx.x;
    if (idx >= H_DIM * H_DIM) return;
    int n = idx >> 7, k = idx & (H_DIM - 1);
    Wt[idx] = (_Float16)W[k * H_DIM + n];  // Wt[n][k] = W[k][n]
}

// ---------------- WMMA GEMM: C(Nx128,f32) = A(Nx128,f16) x B(128x128) ----------------
// Bt (B^T, 128x128 f16) is DMA'd into LDS once per workgroup by the Tensor Data
// Mover (TDM), with pad_enable inserting 16B per 256B row for bank-conflict-free
// ds_load_b128 fragment reads. B fragment layout == A fragment layout (B^T rows
// indexed by N play the role of A rows indexed by M).
__global__ __launch_bounds__(256)
void k_gemm_wmma(const _Float16* __restrict__ A, const _Float16* __restrict__ Bt,
                 float* __restrict__ C) {
    extern __shared__ _Float16 smem[];   // 128 rows x 136 halves (TDM-padded)

    // ---- wave 0: issue TDM descriptor to pull Bt (32KB) into LDS ----
    if (threadIdx.x < 32) {
        unsigned long long ga = (unsigned long long)Bt;
        v4u g0;
        g0[0] = 1u;                                   // count=1 valid descriptor
        g0[1] = 0u;                                   // lds_addr = 0 (dynamic LDS base)
        g0[2] = (unsigned)(ga & 0xFFFFFFFFu);         // global_addr[31:0]
        g0[3] = (unsigned)((ga >> 32) & 0x01FFFFFFu)  // global_addr[56:32]
                | 0x80000000u;                        // type = 2 ("image")
        v8i g1;
        g1[0] = (int)((1u << 16)      // data_size = 1 -> 2 bytes
                    | (1u << 20)      // pad_enable
                    | (5u << 22)      // pad_interval: 64 DW (256B) between pads
                    | (3u << 25));    // pad_amount: 4 DW (16B)
        g1[1] = (int)(128u << 16);    // tensor_dim0 = 128 (low 16 in hi half)
        g1[2] = (int)(128u << 16);    // tensor_dim0 hi=0 | tensor_dim1 = 128 (lo16)
        g1[3] = (int)(128u << 16);    // tensor_dim1 hi=0 | tile_dim0 = 128
        g1[4] = (int)(128u);          // tile_dim1 = 128, tile_dim2 = 0
        g1[5] = 128;                  // tensor_dim0_stride = 128 (lo32)
        g1[6] = 0;                    // stride0 hi | tensor_dim1_stride lo (unused, 2D)
        g1[7] = 0;
        v4i gz  = {0, 0, 0, 0};               // groups 2/3 unused for 2D tensor
        v8i gz8 = {0, 0, 0, 0, 0, 0, 0, 0};   // extra group (6-arg toolchain form)
        __builtin_amdgcn_tensor_load_to_lds(g0, g1, gz, gz, gz8, 0);
        __builtin_amdgcn_s_wait_tensorcnt(0);         // TDM done -> LDS valid
    }
    __syncthreads();                                  // publish to all 8 waves

    const int wave = blockIdx.x * 8 + (threadIdx.x >> 5);
    const int NSTRIPS = N_NODES / 16;          // 6250 strips of 16 rows
    if (wave >= NSTRIPS) return;               // wave-uniform: EXEC all-1s past here
    const int lane = threadIdx.x & 31;
    const int half = lane >> 4;                // K half-group select (ISA 16-bit A layout)
    const int r    = lane & 15;                // M row (A) / N col (B^T)
    const int m0   = wave * 16;

    // A fragments for all 4 K-steps, reused across the 8 N-tiles.
    const _Float16* Arow = A + (size_t)(m0 + r) * H_DIM;
    AFrag fa[4];
#pragma unroll
    for (int kt = 0; kt < 4; ++kt) {
        const int k0 = kt * 32;
        fa[kt].h[0] = *(const v8h*)(Arow + k0 + half * 8);        // K = k0 + half*8 .. +7
        fa[kt].h[1] = *(const v8h*)(Arow + k0 + 16 + half * 8);   // K = k0+16+half*8 .. +7
    }

#pragma unroll
    for (int nt = 0; nt < 8; ++nt) {
        const int n0 = nt * 16;
        const _Float16* Brow = smem + (size_t)(n0 + r) * LDS_ROW_H;
        v8f c = {};
#pragma unroll
        for (int kt = 0; kt < 4; ++kt) {
            const int k0 = kt * 32;
            AFrag fb;
            fb.h[0] = *(const v8h*)(Brow + k0 + half * 8);        // ds_load_b128
            fb.h[1] = *(const v8h*)(Brow + k0 + 16 + half * 8);
            c = __builtin_amdgcn_wmma_f32_16x16x32_f16(
                    false, fa[kt].v, false, fb.v, (short)0, c, false, false);
        }
        // D layout: VGPR rr -> row m0 + half*8 + rr ; col n0 + r
        float* Cc = C + (size_t)(m0 + half * 8) * H_DIM + (n0 + r);
#pragma unroll
        for (int rr = 0; rr < 8; ++rr)
            Cc[(size_t)rr * H_DIM] = c[rr];
    }
}

// ---------------- self-loop init: agg = dinv^2 * t ----------------
__global__ void k_selfloop(const float* __restrict__ t, const float* __restrict__ dinv,
                           float* __restrict__ agg) {
    int idx = blockIdx.x * blockDim.x + threadIdx.x;
    if (idx >= N_NODES * 32) return;
    int i = idx >> 5, f = (idx & 31) * 4;
    float s = dinv[i]; s = s * s;
    const float4 v = *(const float4*)(t + (size_t)i * H_DIM + f);
    float4 o = make_float4(v.x * s, v.y * s, v.z * s, v.w * s);
    *(float4*)(agg + (size_t)i * H_DIM + f) = o;
}

// ---------------- edge scatter: wave per edge, lane = 4 features ----------------
__global__ __launch_bounds__(256)
void k_scatter(const int* __restrict__ ei, const float* __restrict__ dinv,
               const float* __restrict__ t, float* __restrict__ agg) {
    int e = blockIdx.x * 8 + (threadIdx.x >> 5);
    if (e >= E_EDGES) return;
    int s = ei[e];
    int d = ei[E_EDGES + e];
    float w = dinv[s] * dinv[d];
    int f = (threadIdx.x & 31) * 4;
    const float4 v = *(const float4*)(t + (size_t)s * H_DIM + f);
    float* a = agg + (size_t)d * H_DIM + f;
    atomicAdd(a + 0, w * v.x);
    atomicAdd(a + 1, w * v.y);
    atomicAdd(a + 2, w * v.z);
    atomicAdd(a + 3, w * v.w);
}

// ---------------- bias + BN(eval) + ReLU, emit f16 (next GEMM) and f32 ----------------
__global__ void k_bnrelu(const float* __restrict__ agg, const float* __restrict__ b,
                         const float* __restrict__ g, const float* __restrict__ bb,
                         const float* __restrict__ m, const float* __restrict__ v,
                         _Float16* __restrict__ h16, float* __restrict__ h32) {
    int idx = blockIdx.x * blockDim.x + threadIdx.x;
    if (idx >= N_NODES * H_DIM) return;
    int f = idx & (H_DIM - 1);
    float a = agg[idx] + b[f];
    float y = (a - m[f]) * rsqrtf(v[f] + BN_EPS) * g[f] + bb[f];
    y = fmaxf(y, 0.0f);
    h16[idx] = (_Float16)y;
    h32[idx] = y;
}

// ---------------- pooling ----------------
__global__ void k_zero(float* p, int n) {
    int i = blockIdx.x * blockDim.x + threadIdx.x;
    if (i < n) p[i] = 0.0f;
}

__global__ __launch_bounds__(256)
void k_pool(const float* __restrict__ h, const int* __restrict__ batch,
            float* __restrict__ sums, float* __restrict__ cnt) {
    int i = blockIdx.x * 8 + (threadIdx.x >> 5);
    if (i >= N_NODES) return;
    int g = batch[i];
    int lane = threadIdx.x & 31;
    int f = lane * 4;
    const float4 v = *(const float4*)(h + (size_t)i * H_DIM + f);
    float* s = sums + (size_t)g * H_DIM + f;
    atomicAdd(s + 0, v.x);
    atomicAdd(s + 1, v.y);
    atomicAdd(s + 2, v.z);
    atomicAdd(s + 3, v.w);
    if (lane == 0) atomicAdd(cnt + g, 1.0f);
}

__global__ void k_final(const float* __restrict__ sums, const float* __restrict__ cnt,
                        const float* __restrict__ lw, const float* __restrict__ lb,
                        float* __restrict__ out) {
    int t = threadIdx.x;
    if (t >= G_GRAPHS * 2) return;
    int g = t >> 1, c = t & 1;
    float inv = 1.0f / fmaxf(cnt[g], 1.0f);
    float acc = lb[c];
#pragma unroll 4
    for (int f = 0; f < H_DIM; ++f)
        acc += sums[g * H_DIM + f] * inv * lw[f * 2 + c];
    out[t] = acc;
}

// ---------------- host orchestration ----------------
extern "C" void kernel_launch(void* const* d_in, const int* in_sizes, int n_in,
                              void* d_out, int out_size, void* d_ws, size_t ws_size,
                              hipStream_t stream) {
    const float* x     = (const float*)d_in[0];
    const int*   ei    = (const int*)d_in[1];
    const int*   batch = (const int*)d_in[2];
    const float* W1    = (const float*)d_in[3];
    const float* b1    = (const float*)d_in[4];
    const float* W2    = (const float*)d_in[5];
    const float* b2    = (const float*)d_in[6];
    const float* W3    = (const float*)d_in[7];
    const float* b3    = (const float*)d_in[8];
    const float* linW  = (const float*)d_in[9];
    const float* linb  = (const float*)d_in[10];
    const float* bn_g[3] = {(const float*)d_in[11], (const float*)d_in[15], (const float*)d_in[19]};
    const float* bn_b[3] = {(const float*)d_in[12], (const float*)d_in[16], (const float*)d_in[20]};
    const float* bn_m[3] = {(const float*)d_in[13], (const float*)d_in[17], (const float*)d_in[21]};
    const float* bn_v[3] = {(const float*)d_in[14], (const float*)d_in[18], (const float*)d_in[22]};
    const float* layer_b[3] = {b1, b2, b3};
    float* out = (float*)d_out;

    // workspace carve-up (256B aligned)
    char* ws = (char*)d_ws;
    size_t off = 0;
    auto carve = [&](size_t bytes) { char* p = ws + off; off = (off + bytes + 255) & ~(size_t)255; return p; };
    float*    deg   = (float*)   carve((size_t)N_NODES * 4);
    float*    dinv  = (float*)   carve((size_t)N_NODES * 4);
    float*    buf_t = (float*)   carve((size_t)N_NODES * H_DIM * 4);   // GEMM out / h32
    float*    agg   = (float*)   carve((size_t)N_NODES * H_DIM * 4);
    _Float16* h16   = (_Float16*)carve((size_t)N_NODES * H_DIM * 2);
    _Float16* w2t   = (_Float16*)carve((size_t)H_DIM * H_DIM * 2);
    _Float16* w3t   = (_Float16*)carve((size_t)H_DIM * H_DIM * 2);
    float*    sums  = (float*)   carve((size_t)G_GRAPHS * (H_DIM + 1) * 4);  // sums + cnt contiguous
    float*    cnt   = sums + G_GRAPHS * H_DIM;

    const int NH  = N_NODES * H_DIM;                 // 12.8M
    const int g_nh    = (NH + 255) / 256;
    const int g_n     = (N_NODES + 255) / 256;
    const int g_e_thr = (E_EDGES + 255) / 256;
    const int g_n32   = (N_NODES * 32 + 255) / 256;
    const int g_edge  = (E_EDGES + 7) / 8;           // wave per edge, 8 waves/block
    const int g_strip = (N_NODES / 16 + 7) / 8;      // wave per 16-row strip
    const int g_nodeW = (N_NODES + 7) / 8;           // wave per node

    // degree + normalization
    k_init_deg<<<g_n, 256, 0, stream>>>(deg);
    k_deg_edges<<<g_e_thr, 256, 0, stream>>>(ei, deg);
    k_dinv<<<g_n, 256, 0, stream>>>(deg, dinv);

    // f16 transposed weights for WMMA layers
    k_wt16<<<(H_DIM * H_DIM + 255) / 256, 256, 0, stream>>>(W2, w2t);
    k_wt16<<<(H_DIM * H_DIM + 255) / 256, 256, 0, stream>>>(W3, w3t);

    // ---- layer 1 (K=3 scalar GEMM) ----
    k_gemm_in<<<g_nh, 256, 0, stream>>>(x, W1, buf_t);
    k_selfloop<<<g_n32, 256, 0, stream>>>(buf_t, dinv, agg);
    k_scatter<<<g_edge, 256, 0, stream>>>(ei, dinv, buf_t, agg);
    k_bnrelu<<<g_nh, 256, 0, stream>>>(agg, layer_b[0], bn_g[0], bn_b[0], bn_m[0], bn_v[0], h16, buf_t);

    // ---- layer 2 (WMMA GEMM, TDM-staged B) ----
    k_gemm_wmma<<<g_strip, 256, LDS_BYTES, stream>>>(h16, w2t, buf_t);
    k_selfloop<<<g_n32, 256, 0, stream>>>(buf_t, dinv, agg);
    k_scatter<<<g_edge, 256, 0, stream>>>(ei, dinv, buf_t, agg);
    k_bnrelu<<<g_nh, 256, 0, stream>>>(agg, layer_b[1], bn_g[1], bn_b[1], bn_m[1], bn_v[1], h16, buf_t);

    // ---- layer 3 (WMMA GEMM, TDM-staged B) ----
    k_gemm_wmma<<<g_strip, 256, LDS_BYTES, stream>>>(h16, w3t, buf_t);
    k_selfloop<<<g_n32, 256, 0, stream>>>(buf_t, dinv, agg);
    k_scatter<<<g_edge, 256, 0, stream>>>(ei, dinv, buf_t, agg);
    k_bnrelu<<<g_nh, 256, 0, stream>>>(agg, layer_b[2], bn_g[2], bn_b[2], bn_m[2], bn_v[2], h16, buf_t);

    // ---- mean pool + linear ----
    k_zero<<<(G_GRAPHS * (H_DIM + 1) + 255) / 256, 256, 0, stream>>>(sums, G_GRAPHS * (H_DIM + 1));
    k_pool<<<g_nodeW, 256, 0, stream>>>(buf_t, batch, sums, cnt);
    k_final<<<1, 128, 0, stream>>>(sums, cnt, linW, linb, out);
}
